// AttnBlock_56753697849581
// MI455X (gfx1250) — matrix operand order repaired and checked by reading, not compiled, not run on hardware
//
#include <hip/hip_runtime.h>

typedef __attribute__((ext_vector_type(16))) __bf16 v16bf;
typedef __attribute__((ext_vector_type(8)))  __bf16 v8bf;
typedef __attribute__((ext_vector_type(8)))  float  v8f;

#define C_DIM  512
#define HW_DIM 4096
#define B_DIM  4
#define M_DIM  (B_DIM * HW_DIM)

// ---------------------------------------------------------------------------
// WMMA fragment helpers (bf16 16x16x32, wave32)
// A 16x32 (MxK), row-major, lda elements:
//   lane r=lane&15 is row M=r; hf=lane>>4 selects K {0..7,16..23} vs {8..15,24..31}
// B via B^T row-major (row = N column of B), ldb elements:
//   lane r is column N=r; hf selects K {0..7,16..23} vs {8..15,24..31} (same as A)
// C/D 16x16 f32: vgpr v, row M = v + 8*hf, col N = lane&15
// ---------------------------------------------------------------------------
static __device__ __forceinline__ v16bf load_afrag(const __bf16* A, int lda,
                                                   int r, int hf, int koff) {
  const __bf16* p = A + (size_t)r * lda + koff + hf * 8;
  v8bf lo = *(const v8bf*)p;
  v8bf hi = *(const v8bf*)(p + 16);
  v16bf out;
#pragma unroll
  for (int i = 0; i < 8; ++i) { out[i] = lo[i]; out[i + 8] = hi[i]; }
  return out;
}

static __device__ __forceinline__ v16bf load_bfrag(const __bf16* Bt, int ldb,
                                                   int r, int hf, int koff) {
  const __bf16* p = Bt + (size_t)r * ldb + koff + hf * 8;
  v8bf lo = *(const v8bf*)p;
  v8bf hi = *(const v8bf*)(p + 16);
  v16bf out;
#pragma unroll
  for (int i = 0; i < 8; ++i) { out[i] = lo[i]; out[i + 8] = hi[i]; }
  return out;
}

static __device__ __forceinline__ v8f wmma_bf16(v16bf a, v16bf b, v8f c) {
  return __builtin_amdgcn_wmma_f32_16x16x32_bf16(false, a, false, b,
                                                 (short)0, c, false, false);
}

static __device__ __forceinline__ v8f vzero8() {
  v8f z;
#pragma unroll
  for (int i = 0; i < 8; ++i) z[i] = 0.0f;
  return z;
}

// ---------------------------------------------------------------------------
// Kernel 1: GroupNorm (32 groups of 16 channels) -> bf16
// grid: B*32 blocks, 256 threads
// ---------------------------------------------------------------------------
__global__ __launch_bounds__(256) void gn_kernel(const float* __restrict__ x,
                                                 const float* __restrict__ gamma,
                                                 const float* __restrict__ beta,
                                                 __bf16* __restrict__ h) {
  __shared__ float ssum[256];
  __shared__ float ssq[256];
  const int bid = blockIdx.x;
  const int b = bid >> 5, g = bid & 31;
  const int tid = threadIdx.x;
  const size_t base = (size_t)b * HW_DIM * C_DIM + g * 16;

  float s = 0.0f, sq = 0.0f;
  for (int i = tid; i < HW_DIM * 16; i += 256) {
    const int p = i >> 4, c = i & 15;
    const float v = x[base + (size_t)p * C_DIM + c];
    s += v; sq += v * v;
  }
  ssum[tid] = s; ssq[tid] = sq;
  __syncthreads();
  for (int st = 128; st > 0; st >>= 1) {
    if (tid < st) { ssum[tid] += ssum[tid + st]; ssq[tid] += ssq[tid + st]; }
    __syncthreads();
  }
  const float mean = ssum[0] * (1.0f / 65536.0f);
  const float var  = ssq[0] * (1.0f / 65536.0f) - mean * mean;
  const float rinv = rsqrtf(var + 1e-5f);

  for (int i = tid; i < HW_DIM * 16; i += 256) {
    const int p = i >> 4, c = i & 15;
    const size_t idx = base + (size_t)p * C_DIM + c;
    const float v = (x[idx] - mean) * rinv * gamma[g * 16 + c] + beta[g * 16 + c];
    h[idx] = (__bf16)v;
  }
}

// ---------------------------------------------------------------------------
// Kernel 2: transpose + fp32->bf16 convert the four 512x512 weights
// grid: 1024 blocks, 256 threads (one element of each matrix per thread)
// ---------------------------------------------------------------------------
__global__ void transpose_w(const float* __restrict__ wq, const float* __restrict__ wk,
                            const float* __restrict__ wv, const float* __restrict__ wp,
                            __bf16* __restrict__ wqT, __bf16* __restrict__ wkT,
                            __bf16* __restrict__ wvT, __bf16* __restrict__ wpT) {
  const int idx = blockIdx.x * 256 + threadIdx.x;   // over 512*512
  const int n = idx >> 9, k = idx & 511;
  const size_t src = (size_t)k * C_DIM + n;
  const size_t dst = (size_t)n * C_DIM + k;
  wqT[dst] = (__bf16)wq[src];
  wkT[dst] = (__bf16)wk[src];
  wvT[dst] = (__bf16)wv[src];
  wpT[dst] = (__bf16)wp[src];
}

// ---------------------------------------------------------------------------
// Kernel 3: GEMM out = h @ W + b  (M=16384, N=512, K=512), bf16 in/out
// block = 256 thr = 8 waves in 2x4 grid; wave tile 32x32; block tile 64x128
// grid = (M/64, N/128) = (256, 4). transposed!=0 stores V as (b, c, p).
// ---------------------------------------------------------------------------
__global__ __launch_bounds__(256) void gemm_qkv(const __bf16* __restrict__ A,
                                                const __bf16* __restrict__ Bt,
                                                const float* __restrict__ bias,
                                                __bf16* __restrict__ out,
                                                int transposed) {
  const int tid = threadIdx.x;
  const int w = tid >> 5, lane = tid & 31;
  const int hf = lane >> 4, r = lane & 15;
  const int wm = w >> 2, wn = w & 3;
  const int mbase = blockIdx.x * 64 + wm * 32;
  const int nbase = blockIdx.y * 128 + wn * 32;

  v8f acc[2][2];
#pragma unroll
  for (int mi = 0; mi < 2; ++mi)
#pragma unroll
    for (int ni = 0; ni < 2; ++ni) acc[mi][ni] = vzero8();

  const __bf16* A0 = A + (size_t)mbase * C_DIM;
  const __bf16* B0 = Bt + (size_t)nbase * C_DIM;

#pragma unroll
  for (int kk = 0; kk < 16; ++kk) {
    const int koff = kk * 32;
    v16bf a0 = load_afrag(A0, C_DIM, r, hf, koff);
    v16bf a1 = load_afrag(A0 + 16 * C_DIM, C_DIM, r, hf, koff);
    v16bf b0 = load_bfrag(B0, C_DIM, r, hf, koff);
    v16bf b1 = load_bfrag(B0 + 16 * C_DIM, C_DIM, r, hf, koff);
    acc[0][0] = wmma_bf16(a0, b0, acc[0][0]);
    acc[0][1] = wmma_bf16(a0, b1, acc[0][1]);
    acc[1][0] = wmma_bf16(a1, b0, acc[1][0]);
    acc[1][1] = wmma_bf16(a1, b1, acc[1][1]);
  }

#pragma unroll
  for (int mi = 0; mi < 2; ++mi)
#pragma unroll
    for (int ni = 0; ni < 2; ++ni) {
      const int col = nbase + ni * 16 + r;
      const float bv = bias[col];
#pragma unroll
      for (int v = 0; v < 8; ++v) {
        const int grow = mbase + mi * 16 + v + 8 * hf;
        const float val = acc[mi][ni][v] + bv;
        if (transposed) {
          const int bb = grow >> 12;            // batch
          const int p  = grow & (HW_DIM - 1);   // pixel
          out[((size_t)bb * C_DIM + col) * HW_DIM + p] = (__bf16)val;
        } else {
          out[(size_t)grow * C_DIM + col] = (__bf16)val;
        }
      }
    }
}

// ---------------------------------------------------------------------------
// Kernel 4: fused attention + output projection + residual
// block = 16 queries, 256 threads = 8 waves; wave w owns key-cols [w*64,w*64+64)
// Streams keys in tiles of 512 with online (flash) softmax; scores in LDS.
// grid = B * (HW/16) = 1024 blocks
// ---------------------------------------------------------------------------
__global__ __launch_bounds__(256) void attn_kernel(const __bf16* __restrict__ q,
                                                   const __bf16* __restrict__ kbuf,
                                                   const __bf16* __restrict__ vT,
                                                   const __bf16* __restrict__ wpT,
                                                   const float* __restrict__ bp,
                                                   const float* __restrict__ x,
                                                   float* __restrict__ out) {
  __shared__ float S[16][512];                       // 32 KB scores (fp32)
  __shared__ __align__(16) __bf16 P[16][512];        // 16 KB probs / O exchange
  __shared__ float red[16][16];
  __shared__ float mrow[16], lrow[16], arow[16];

  const int tid = threadIdx.x;
  const int w = tid >> 5, lane = tid & 31;
  const int hf = lane >> 4, r = lane & 15;
  const int b = blockIdx.x >> 8;
  const int q0 = (blockIdx.x & 255) * 16;
  const float scale = 0.044194173824159216f;         // 512^-0.5

  const __bf16* Qb = q + ((size_t)b * HW_DIM + q0) * C_DIM;

  v8f o[4];
#pragma unroll
  for (int nf = 0; nf < 4; ++nf) o[nf] = vzero8();
  if (tid < 16) { mrow[tid] = -3.0e38f; lrow[tid] = 0.0f; }
  __syncthreads();

  for (int kb = 0; kb < 8; ++kb) {
    const int key0 = kb * 512;

    // ---- S = Q @ K^T (this wave: 16 x 64 keys) ----
    v8f s4[4];
#pragma unroll
    for (int nf = 0; nf < 4; ++nf) s4[nf] = vzero8();
    const __bf16* Kb = kbuf + ((size_t)b * HW_DIM + key0 + w * 64) * C_DIM;
#pragma unroll
    for (int kk = 0; kk < 16; ++kk) {
      v16bf aq = load_afrag(Qb, C_DIM, r, hf, kk * 32);
#pragma unroll
      for (int nf = 0; nf < 4; ++nf) {
        v16bf bk = load_bfrag(Kb + (size_t)nf * 16 * C_DIM, C_DIM, r, hf, kk * 32);
        s4[nf] = wmma_bf16(aq, bk, s4[nf]);
      }
    }
#pragma unroll
    for (int nf = 0; nf < 4; ++nf)
#pragma unroll
      for (int v = 0; v < 8; ++v)
        S[v + 8 * hf][w * 64 + nf * 16 + r] = s4[nf][v] * scale;
    __syncthreads();

    // ---- online softmax over this 512-key tile ----
    const int rr = tid >> 4, j = tid & 15, cbase = j * 32;
    float mx = -3.0e38f;
    for (int c = 0; c < 32; ++c) mx = fmaxf(mx, S[rr][cbase + c]);
    red[rr][j] = mx;
    __syncthreads();
    if (tid < 16) {
      float mnew = mrow[tid];
      for (int jj = 0; jj < 16; ++jj) mnew = fmaxf(mnew, red[tid][jj]);
      arow[tid] = __expf(mrow[tid] - mnew);
      mrow[tid] = mnew;
    }
    __syncthreads();
    const float mr = mrow[rr];
    float sum = 0.0f;
    for (int c = 0; c < 32; ++c) {
      const float e = __expf(S[rr][cbase + c] - mr);
      P[rr][cbase + c] = (__bf16)e;
      sum += e;
    }
    red[rr][j] = sum;
    __syncthreads();
    if (tid < 16) {
      float s = 0.0f;
      for (int jj = 0; jj < 16; ++jj) s += red[tid][jj];
      lrow[tid] = lrow[tid] * arow[tid] + s;
    }
    __syncthreads();

    // ---- rescale O, then O += P @ V_tile ----
#pragma unroll
    for (int nf = 0; nf < 4; ++nf)
#pragma unroll
      for (int v = 0; v < 8; ++v) o[nf][v] *= arow[v + 8 * hf];

    const __bf16* Vb = vT + ((size_t)b * C_DIM + w * 64) * HW_DIM + key0;
#pragma unroll
    for (int kk = 0; kk < 16; ++kk) {
      v16bf ap = load_afrag(&P[0][0], 512, r, hf, kk * 32);
#pragma unroll
      for (int nf = 0; nf < 4; ++nf) {
        v16bf bv = load_bfrag(Vb + (size_t)nf * 16 * HW_DIM, HW_DIM, r, hf, kk * 32);
        o[nf] = wmma_bf16(ap, bv, o[nf]);
      }
    }
    __syncthreads();   // all P reads done before next tile rewrites S/P/arow
  }

  // ---- normalize and exchange O through LDS (reuse P) ----
  if (tid < 16) arow[tid] = 1.0f / lrow[tid];
  __syncthreads();
#pragma unroll
  for (int nf = 0; nf < 4; ++nf)
#pragma unroll
    for (int v = 0; v < 8; ++v) {
      o[nf][v] *= arow[v + 8 * hf];
      P[v + 8 * hf][w * 64 + nf * 16 + r] = (__bf16)o[nf][v];
    }
  __syncthreads();

  // ---- proj: out = O @ wp + bp + x ----
  v8f pa[4];
#pragma unroll
  for (int nf = 0; nf < 4; ++nf) pa[nf] = vzero8();
  const __bf16* Wb = wpT + (size_t)(w * 64) * C_DIM;
#pragma unroll
  for (int kk = 0; kk < 16; ++kk) {
    v16bf ao = load_afrag(&P[0][0], 512, r, hf, kk * 32);
#pragma unroll
    for (int nf = 0; nf < 4; ++nf) {
      v16bf bw = load_bfrag(Wb + (size_t)nf * 16 * C_DIM, C_DIM, r, hf, kk * 32);
      pa[nf] = wmma_bf16(ao, bw, pa[nf]);
    }
  }
#pragma unroll
  for (int nf = 0; nf < 4; ++nf) {
    const int d = w * 64 + nf * 16 + r;
    const float bias = bp[d];
#pragma unroll
    for (int v = 0; v < 8; ++v) {
      const int m = v + 8 * hf;
      const size_t idx = ((size_t)b * HW_DIM + q0 + m) * C_DIM + d;
      out[idx] = pa[nf][v] + bias + x[idx];
    }
  }
}

// ---------------------------------------------------------------------------
// Host launcher
// ---------------------------------------------------------------------------
extern "C" void kernel_launch(void* const* d_in, const int* in_sizes, int n_in,
                              void* d_out, int out_size, void* d_ws, size_t ws_size,
                              hipStream_t stream) {
  const float* x     = (const float*)d_in[0];
  const float* gamma = (const float*)d_in[1];
  const float* beta  = (const float*)d_in[2];
  const float* wq    = (const float*)d_in[3];
  const float* bq    = (const float*)d_in[4];
  const float* wk    = (const float*)d_in[5];
  const float* bk    = (const float*)d_in[6];
  const float* wv    = (const float*)d_in[7];
  const float* bv    = (const float*)d_in[8];
  const float* wp    = (const float*)d_in[9];
  const float* bp    = (const float*)d_in[10];
  float* out = (float*)d_out;

  const size_t MC = (size_t)M_DIM * C_DIM;
  const size_t CC = (size_t)C_DIM * C_DIM;
  __bf16* ws  = (__bf16*)d_ws;
  __bf16* h   = ws;          // M x C  bf16 normalized activations
  __bf16* wqT = h + MC;      // C x C  each, transposed bf16 weights
  __bf16* wkT = wqT + CC;
  __bf16* wvT = wkT + CC;
  __bf16* wpT = wvT + CC;
  __bf16* qb  = wpT + CC;    // M x C
  __bf16* kb  = qb + MC;     // M x C
  __bf16* vT  = kb + MC;     // B x C x HW (transposed V)

  gn_kernel<<<B_DIM * 32, 256, 0, stream>>>(x, gamma, beta, h);
  transpose_w<<<(C_DIM * C_DIM) / 256, 256, 0, stream>>>(wq, wk, wv, wp,
                                                         wqT, wkT, wvT, wpT);
  dim3 ggrid(M_DIM / 64, C_DIM / 128);
  gemm_qkv<<<ggrid, 256, 0, stream>>>(h, wqT, bq, qb, 0);
  gemm_qkv<<<ggrid, 256, 0, stream>>>(h, wkT, bk, kb, 0);
  gemm_qkv<<<ggrid, 256, 0, stream>>>(h, wvT, bv, vT, 1);
  attn_kernel<<<B_DIM * (HW_DIM / 16), 256, 0, stream>>>(qb, kb, vT, wpT, bp, x, out);
}
